// MultiheadedAttention_13194139533572
// MI455X (gfx1250) — matrix operand order repaired and checked by reading, not compiled, not run on hardware
//
#include <hip/hip_runtime.h>
#include <hip/hip_bf16.h>
#include <math.h>

#define B_  4
#define N_  2048
#define DM_ 1024
#define H_  16
#define DH_ 64

#define WSTRIDE 1032   // W LDS col stride (elems): bank-conflict-free B-frags
#define KSTRIDE 72     // K/V tile LDS row stride (elems): via TDM padding
#define VTSTRIDE 40    // transposed-V LDS row stride (elems)
#define PSTRIDE 40     // P tile LDS row stride (elems)
#define SLABS 8        // 128-row slabs per projection block

typedef __bf16 bf16;
typedef __attribute__((ext_vector_type(16))) __bf16 v16bf;
typedef __attribute__((ext_vector_type(8)))  __bf16 v8bf;
typedef __attribute__((ext_vector_type(8)))  float  v8f;
typedef __attribute__((ext_vector_type(4)))  unsigned int v4u;
typedef __attribute__((ext_vector_type(4)))  int v4i;
typedef __attribute__((ext_vector_type(8)))  int v8i;

static __device__ __forceinline__ bf16 f2bf(float f) { return (bf16)f; }

static __device__ __forceinline__ float fast_exp2(float x) {
#if __has_builtin(__builtin_amdgcn_exp2f)
  return __builtin_amdgcn_exp2f(x);
#else
  return exp2f(x);
#endif
}

// ---------------------------------------------------------------------------
// TDM: TENSOR_LOAD_TO_LDS of a 32x64 bf16 tile (row stride 64 elems in memory),
// with LDS padding: 16 bytes inserted every 128 bytes (pad_interval code 4,
// pad_amount code 3) -> LDS rows land at stride KSTRIDE=72 elems, making the
// subsequent ds_load_b128 B-fragment reads bank-conflict-free.
// Descriptor per CDNA5 ISA.
// ---------------------------------------------------------------------------
static __device__ __forceinline__ void tdm_load_tile_bf16(
    unsigned lds_off, const void* gptr)
{
  const unsigned long long ga = (unsigned long long)gptr;

  v4u g0 = (v4u)0u;
  g0[0] = 1u;                                   // count=1, user descriptor
  g0[1] = lds_off;                              // LDS byte address
  g0[2] = (unsigned)(ga & 0xffffffffu);         // global_addr[31:0]
  g0[3] = (unsigned)((ga >> 32) & 0x1ffffffu)   // global_addr[56:32]
        | (2u << 30);                           // type = 2 ("image")

  v8i g1 = (v8i)0;
  g1[0] = (int)((1u << 16)                      // data_size = 1 (2 bytes)
              | (1u << 20)                      // pad_enable
              | (4u << 22)                      // pad_interval: every 32 DWORDs
              | (3u << 25));                    // pad_amount: 4 DWORDs (16 B)
  g1[1] = (int)(64u << 16);                     // tensor_dim0 = 64   (bits 79:48)
  g1[2] = (int)(((unsigned)N_ & 0xffffu) << 16);// tensor_dim1 = N_   (bits 111:80)
  g1[3] = (int)(64u << 16);                     // tile_dim0 = 64     (bits 127:112)
  g1[4] = 32;                                   // tile_dim1 = 32     (bits 143:128)
  g1[5] = 64;                                   // tensor_dim0_stride (bits 207:160)

  v4i z4 = (v4i)0;
#if defined(__clang_major__) && (__clang_major__ >= 23)
  v8i z8 = (v8i)0;
  __builtin_amdgcn_tensor_load_to_lds(g0, g1, z4, z4, z8, 0);
#else
  __builtin_amdgcn_tensor_load_to_lds(g0, g1, z4, z4, 0);
#endif
}

// ---------------------------------------------------------------------------
// Kernel 1: per-head QKV projection.  grid = (B*N/1024, H, 3), block = 256.
// Entire W[h] (64 cols x 1024 k, bf16, transposed, padded stride) staged once
// in dynamic LDS (~129 KB of the 320 KB WGP pool), then 8 row-slabs of 128
// stream through with no further barriers.  Softmax scale * log2(e) folded
// into Q so the attention kernel can use raw v_exp_f32 (exp2).
// ---------------------------------------------------------------------------
__global__ __launch_bounds__(256)
void qkv_proj_kernel(const float* __restrict__ Xq, const float* __restrict__ Xk,
                     const float* __restrict__ Xv,
                     const float* __restrict__ Wq, const float* __restrict__ Wk,
                     const float* __restrict__ Wv,
                     const float* __restrict__ bq, const float* __restrict__ bk,
                     const float* __restrict__ bv,
                     bf16* __restrict__ Qo, bf16* __restrict__ Ko, bf16* __restrict__ Vo)
{
  const int h     = blockIdx.y;
  const int which = blockIdx.z;
  const float* __restrict__ X  = (which == 0) ? Xq : (which == 1) ? Xk : Xv;
  const float* __restrict__ W  = (which == 0) ? Wq : (which == 1) ? Wk : Wv;
  const float* __restrict__ bb = (which == 0) ? bq : (which == 1) ? bk : bv;
  bf16* __restrict__ O         = (which == 0) ? Qo : (which == 1) ? Ko : Vo;
  // 1/sqrt(64) * log2(e) folded into Q; K,V unscaled
  const float scale = (which == 0) ? 0.125f * 1.44269504088896340736f : 1.0f;

  extern __shared__ __align__(32) bf16 Wt[];   // [col][k], stride WSTRIDE

  const int tid  = threadIdx.x;
  const int wave = tid >> 5;
  const int lane = tid & 31;
  const int hlf  = lane >> 4;
  const int l16  = lane & 15;

  // stage entire W[h] transposed as bf16 (65536 elements, 256/thread)
  for (int j = 0; j < (DH_ * DM_) / 256; ++j) {
    int idx = tid + j * 256;
    int k   = idx >> 6;
    int col = idx & 63;
    Wt[col * WSTRIDE + k] = f2bf(W[(h * DM_ + k) * DH_ + col]);
  }
  __syncthreads();

  const int rowBase = blockIdx.x * (128 * SLABS);

  for (int sl = 0; sl < SLABS; ++sl) {
    const int g0   = rowBase + sl * 128 + wave * 16;
    const int rowA = g0 + l16;

    v8f acc[4];
    #pragma unroll
    for (int t = 0; t < 4; ++t)
      #pragma unroll
      for (int i = 0; i < 8; ++i) acc[t][i] = 0.0f;

    #pragma unroll 4
    for (int kk = 0; kk < DM_; kk += 32) {
      // A fragment (16x32 bf16) from global fp32 X
      v16bf a;
      const float* px = X + (size_t)rowA * DM_ + kk;
      #pragma unroll
      for (int i = 0; i < 8; ++i) {
        a[i]     = f2bf(px[hlf * 16 + i]);
        a[i + 8] = f2bf(px[hlf * 16 + 8 + i]);
      }
      #pragma unroll
      for (int t = 0; t < 4; ++t) {
        const v16bf bfrag =
            *(const v16bf*)(&Wt[(t * 16 + l16) * WSTRIDE + kk + hlf * 16]);
        acc[t] = __builtin_amdgcn_wmma_f32_16x16x32_bf16(
            false, a, false, bfrag, (short)0, acc[t], false, false);
      }
    }

    // bias + scale + bf16 store to [B,H,N,DH]
    #pragma unroll
    for (int t = 0; t < 4; ++t) {
      const float bias = bb[h * DH_ + t * 16 + l16];
      #pragma unroll
      for (int r = 0; r < 8; ++r) {
        int g    = g0 + r + 8 * hlf;     // C/D layout: M = r + 8*half
        int bidx = g >> 11;              // / N_
        int n    = g & (N_ - 1);
        float v  = (acc[t][r] + bias) * scale;
        O[(((size_t)bidx * H_ + h) * N_ + n) * DH_ + t * 16 + l16] = f2bf(v);
      }
    }
  }
}

// ---------------------------------------------------------------------------
// Kernel 2: causal flash attention over bf16 Q/K/V.
// grid = (N/128, H, B), block = 256 (8 waves; 16 q-rows per wave).
// K/V tiles double-buffered via Tensor Data Mover: tile kb+1 issued while
// tile kb is consumed; s_wait_tensorcnt 2 (in-order completion) + barrier.
// ---------------------------------------------------------------------------
__global__ __launch_bounds__(256)
void flash_attn_kernel(const bf16* __restrict__ Q, const bf16* __restrict__ K,
                       const bf16* __restrict__ V, float* __restrict__ out)
{
  const int bidx = blockIdx.z;
  const int h    = blockIdx.y;
  const int q0   = blockIdx.x * 128;

  const int tid  = threadIdx.x;
  const int wave = tid >> 5;
  const int lane = tid & 31;
  const int hlf  = lane >> 4;
  const int l16  = lane & 15;

  const int qw   = q0 + wave * 16;   // wave's q-tile base row
  const int qmax = qw + 15;

  __shared__ __align__(32) bf16 Kt [2][32 * KSTRIDE];   // TDM, padded rows
  __shared__ __align__(32) bf16 Vst[2][32 * KSTRIDE];   // TDM, padded rows
  __shared__ __align__(32) bf16 Vt [DH_ * VTSTRIDE];    // [d][key], padded
  __shared__ __align__(32) bf16 Pt [8 * 16 * PSTRIDE];  // per-wave P, padded

  const size_t headBase = ((size_t)bidx * H_ + h) * N_ * DH_;
  const bf16* __restrict__ Qb = Q + headBase;
  const bf16* __restrict__ Kb = K + headBase;
  const bf16* __restrict__ Vb = V + headBase;

  const unsigned ldsK[2] = { (unsigned)(unsigned long long)(void*)&Kt[0][0],
                             (unsigned)(unsigned long long)(void*)&Kt[1][0] };
  const unsigned ldsV[2] = { (unsigned)(unsigned long long)(void*)&Vst[0][0],
                             (unsigned)(unsigned long long)(void*)&Vst[1][0] };

  // Q A-fragments for this wave's 16 rows (K-dim = d, two 32-chunks)
  v16bf qf[2];
  {
    const bf16* pq = Qb + (size_t)(qw + l16) * DH_;
    #pragma unroll
    for (int c = 0; c < 2; ++c) {
      v8bf lo = *(const v8bf*)(pq + c * 32 + hlf * 16);
      v8bf hi = *(const v8bf*)(pq + c * 32 + hlf * 16 + 8);
      #pragma unroll
      for (int i = 0; i < 8; ++i) { qf[c][i] = lo[i]; qf[c][i + 8] = hi[i]; }
    }
  }

  v8f o[4];
  #pragma unroll
  for (int t = 0; t < 4; ++t)
    #pragma unroll
    for (int i = 0; i < 8; ++i) o[t][i] = 0.0f;
  float m[8], lsum[8];
  #pragma unroll
  for (int r = 0; r < 8; ++r) { m[r] = -1e30f; lsum[r] = 0.0f; }

  bf16* myP = Pt + wave * (16 * PSTRIDE);
  const int kbmax = (q0 + 127) >> 5;

  // prologue: issue TDM for tile 0 into buffer 0
  if (wave == 0) {
    tdm_load_tile_bf16(ldsK[0], Kb);
    tdm_load_tile_bf16(ldsV[0], Vb);
  }

  for (int kb = 0; kb <= kbmax; ++kb) {
    const int buf = kb & 1;

    // issue next tile into the other buffer, then wait for current tile.
    // TDM ops complete in order: <=2 outstanding means tile kb has landed.
    if (wave == 0) {
      if (kb < kbmax) {
        tdm_load_tile_bf16(ldsK[buf ^ 1], Kb + (size_t)(kb + 1) * 32 * DH_);
        tdm_load_tile_bf16(ldsV[buf ^ 1], Vb + (size_t)(kb + 1) * 32 * DH_);
        __builtin_amdgcn_s_wait_tensorcnt(2);
      } else {
        __builtin_amdgcn_s_wait_tensorcnt(0);
      }
    }
    __syncthreads();

    // ---- transpose V staging -> Vt[d][key] (LDS -> LDS) ----
    #pragma unroll
    for (int j = 0; j < 8; ++j) {
      int idx  = tid + j * 256;          // 2048 elements
      int keyl = idx >> 6;
      int d    = idx & 63;
      Vt[d * VTSTRIDE + keyl] = Vst[buf][keyl * KSTRIDE + d];
    }
    __syncthreads();

    if (kb * 32 <= qmax) {
      // ---- S = Q K^T : two 16-key subtiles, K B-frags from padded LDS ----
      v8f s[2];
      #pragma unroll
      for (int kt = 0; kt < 2; ++kt) {
        #pragma unroll
        for (int i = 0; i < 8; ++i) s[kt][i] = 0.0f;
        const int keyl = kt * 16 + l16;            // B-frag column (tile-local)
        #pragma unroll
        for (int c = 0; c < 2; ++c) {
          const v16bf kf =
              *(const v16bf*)(&Kt[buf][keyl * KSTRIDE + c * 32 + hlf * 16]);
          s[kt] = __builtin_amdgcn_wmma_f32_16x16x32_bf16(
              false, qf[c], false, kf, (short)0, s[kt], false, false);
        }
      }

      // ---- causal mask (only tiles touching the diagonal) ----
      if (kb * 32 + 31 > qw) {
        #pragma unroll
        for (int kt = 0; kt < 2; ++kt) {
          const int key = kb * 32 + kt * 16 + l16;
          #pragma unroll
          for (int r = 0; r < 8; ++r) {
            const int row = qw + r + 8 * hlf;
            if (key > row) s[kt][r] = -1e30f;
          }
        }
      }

      // ---- online softmax in log2 domain (scale*log2e folded into Q) ----
      #pragma unroll
      for (int r = 0; r < 8; ++r) {
        float mx = fmaxf(s[0][r], s[1][r]);
        #pragma unroll
        for (int off = 1; off < 16; off <<= 1)
          mx = fmaxf(mx, __shfl_xor(mx, off, 32));
        const float mnew  = fmaxf(m[r], mx);
        const float alpha = fast_exp2(m[r] - mnew);
        m[r] = mnew;
        const float p0 = fast_exp2(s[0][r] - mnew);
        const float p1 = fast_exp2(s[1][r] - mnew);
        s[0][r] = p0; s[1][r] = p1;
        float rs = p0 + p1;
        #pragma unroll
        for (int off = 1; off < 16; off <<= 1)
          rs += __shfl_xor(rs, off, 32);
        lsum[r] = lsum[r] * alpha + rs;
        #pragma unroll
        for (int t = 0; t < 4; ++t) o[t][r] *= alpha;
      }

      // ---- P: accumulator layout -> A-fragment layout via LDS ----
      #pragma unroll
      for (int r = 0; r < 8; ++r) {
        const int row = r + 8 * hlf;
        myP[row * PSTRIDE + l16]      = f2bf(s[0][r]);
        myP[row * PSTRIDE + 16 + l16] = f2bf(s[1][r]);
      }
      v16bf pf;
      {
        const bf16* pp = myP + l16 * PSTRIDE;
        v8bf lo = *(const v8bf*)(pp + hlf * 16);
        v8bf hi = *(const v8bf*)(pp + hlf * 16 + 8);
        #pragma unroll
        for (int i = 0; i < 8; ++i) { pf[i] = lo[i]; pf[i + 8] = hi[i]; }
      }

      // ---- O += P V : 4 d-tiles, conflict-free B-frags from Vt ----
      #pragma unroll
      for (int t = 0; t < 4; ++t) {
        const v16bf vf =
            *(const v16bf*)(&Vt[(t * 16 + l16) * VTSTRIDE + hlf * 16]);
        o[t] = __builtin_amdgcn_wmma_f32_16x16x32_bf16(
            false, pf, false, vf, (short)0, o[t], false, false);
      }
    }
    __syncthreads();
  }

  // ---- normalize + store fp32, heads concatenated along feature dim ----
  #pragma unroll
  for (int r = 0; r < 8; ++r) {
    const int row = qw + r + 8 * hlf;
    const float inv = 1.0f / lsum[r];
    #pragma unroll
    for (int t = 0; t < 4; ++t) {
      out[((size_t)bidx * N_ + row) * DM_ + h * DH_ + t * 16 + l16] =
          o[t][r] * inv;
    }
  }
}

// ---------------------------------------------------------------------------
extern "C" void kernel_launch(void* const* d_in, const int* in_sizes, int n_in,
                              void* d_out, int out_size, void* d_ws, size_t ws_size,
                              hipStream_t stream) {
  const float* Xq = (const float*)d_in[0];
  const float* Xk = (const float*)d_in[1];
  const float* Xv = (const float*)d_in[2];
  const float* Wq = (const float*)d_in[3];
  const float* Wk = (const float*)d_in[4];
  const float* Wv = (const float*)d_in[5];
  const float* bq = (const float*)d_in[6];
  const float* bk = (const float*)d_in[7];
  const float* bv = (const float*)d_in[8];
  float* out = (float*)d_out;

  const size_t per = (size_t)B_ * H_ * N_ * DH_;   // elements per Q/K/V buffer
  if (ws_size < 3 * per * sizeof(bf16)) return;    // need ~50 MB scratch
  bf16* Qw = (bf16*)d_ws;
  bf16* Kw = Qw + per;
  bf16* Vw = Kw + per;

  const size_t wlds = (size_t)DH_ * WSTRIDE * sizeof(bf16);  // ~129 KB dynamic LDS
  dim3 gProj((B_ * N_) / (128 * SLABS), H_, 3);
  qkv_proj_kernel<<<gProj, 256, wlds, stream>>>(Xq, Xk, Xv, Wq, Wk, Wv,
                                                bq, bk, bv, Qw, Kw, Vw);

  dim3 gAttn(N_ / 128, H_, B_);
  flash_attn_kernel<<<gAttn, 256, 0, stream>>>(Qw, Kw, Vw, out);
}
